// Encoder_1108101562637
// MI455X (gfx1250) — matrix operand order repaired and checked by reading, not compiled, not run on hardware
//
#include <hip/hip_runtime.h>

#define Hd   256
#define Td   256
#define Bd   512
#define Zd   128

typedef __attribute__((ext_vector_type(16))) __bf16 v16bf;
typedef __attribute__((ext_vector_type(8)))  __bf16 v8bf;
typedef __attribute__((ext_vector_type(8)))  float  v8f;

__device__ __forceinline__ __bf16 f2bf(float f) {
    union { float f; unsigned u; } in; in.f = f;
    unsigned u = in.u;
    u += 0x7FFFu + ((u >> 16) & 1u);          // round to nearest even
    unsigned short s = (unsigned short)(u >> 16);
    union { unsigned short s; __bf16 b; } out; out.s = s;
    return out.b;
}
__device__ __forceinline__ float bf2f(__bf16 b) {
    union { unsigned short s; __bf16 b; } in; in.b = b;
    union { unsigned u; float f; } out; out.u = ((unsigned)in.s) << 16;
    return out.f;
}
__device__ __forceinline__ float sigm(float x) { return 1.f / (1.f + __expf(-x)); }
__device__ __forceinline__ float tanh_f(float x) {
    x = fminf(fmaxf(x, -20.f), 20.f);
    float e = __expf(2.f * x);
    return (e - 1.f) / (e + 1.f);
}

// ---------------------------------------------------------------------------
// Swizzle f32 weight [N=1024, Kdim] row-major into bf16 WMMA-B fragments.
// Fragment f = ntile*kcPer + kc holds B-tile (K = kc*32..+31, N = ntile*16..+15).
// Per-lane layout (16-bit B): N = lane%16, element e<8: K = (lane/16)*8+e,
// e>=8: K = 16+(lane/16)*8+(e-8).
// ---------------------------------------------------------------------------
__global__ void swizzleB_kernel(const float* __restrict__ src, __bf16* __restrict__ dst,
                                int Kdim, int kcPer, int total) {
    int i = blockIdx.x * blockDim.x + threadIdx.x;
    if (i >= total) return;
    int e    = i & 15;
    int lane = (i >> 4) & 31;
    int frag = i >> 9;
    int ntile = frag / kcPer;
    int kc    = frag - ntile * kcPer;
    int n = ntile * 16 + (lane & 15);
    int k = kc * 32 + (lane >> 4) * 8 + (e & 7) + ((e >> 3) << 4);
    dst[i] = f2bf(src[(size_t)n * Kdim + k]);
}

// ---------------------------------------------------------------------------
// Layer-0 BiLSTM scan. grid(32,2). 8 waves; wave w owns gate columns
// {g*256 + 32w + [0,32)} for g=i,f,g,o. Recurrent [16x256]x[256x1024] per step.
// B-fragments: kc 0..3 register-resident across the scan (256 VGPR/wave),
// kc 4..7 streamed from L2 each step (anti-hoisted via opaque zero).
// ---------------------------------------------------------------------------
__global__ void __launch_bounds__(256, 1) lstm_l0_kernel(
    const float* __restrict__ x,
    const float* __restrict__ Wih_fw, const float* __restrict__ bih_fw, const float* __restrict__ bhh_fw,
    const float* __restrict__ Wih_bw, const float* __restrict__ bih_bw, const float* __restrict__ bhh_bw,
    const __bf16* __restrict__ WhhS_fw, const __bf16* __restrict__ WhhS_bw,
    __bf16* __restrict__ y0)
{
    __shared__ alignas(16) __bf16 hlds[16 * 256];
    __shared__ float xl[80];

    const int dir   = blockIdx.y;
    const int bbase = blockIdx.x * 16;
    const int tid   = threadIdx.x;
    const int wv    = tid >> 5;
    const int lane  = tid & 31;
    const int hf    = lane >> 4;
    const int lrow  = lane & 15;

    const float*  Wih  = dir ? Wih_bw  : Wih_fw;
    const float*  bih  = dir ? bih_bw  : bih_fw;
    const float*  bhh  = dir ? bhh_bw  : bhh_fw;
    const __bf16* WhhS = dir ? WhhS_bw : WhhS_fw;

    float bsum[8], wih[8][5];
    int   nt[8];
    #pragma unroll
    for (int s = 0; s < 8; ++s) {
        int col = ((s >> 1) << 8) + (wv << 5) + ((s & 1) << 4) + lrow;
        nt[s]   = 16 * (s >> 1) + 2 * wv + (s & 1);
        bsum[s] = bih[col] + bhh[col];
        #pragma unroll
        for (int j = 0; j < 5; ++j) wih[s][j] = Wih[col * 5 + j];
    }

    // register-resident half of Whh fragments (kc 0..3)
    v16bf bReg[4][8];
    #pragma unroll
    for (int kc = 0; kc < 4; ++kc)
        #pragma unroll
        for (int s = 0; s < 8; ++s)
            bReg[kc][s] = *(const v16bf*)&WhhS[((size_t)(nt[s] * 8 + kc)) * 512 + lane * 16];

    for (int i = tid; i < 16 * 256; i += 256) hlds[i] = f2bf(0.f);
    float cst[2][8];
    #pragma unroll
    for (int j = 0; j < 2; ++j)
        #pragma unroll
        for (int r = 0; r < 8; ++r) cst[j][r] = 0.f;
    __syncthreads();

    #pragma unroll 1
    for (int t = 0; t < Td; ++t) {
        const int tt = dir ? (Td - 1 - t) : t;
        if (tid < 80)
            xl[tid] = x[(size_t)(bbase + tid / 5) * Td * 5 + (size_t)tt * 5 + (tid % 5)];
        __syncthreads();   // prev hlds writes + xl visible

        // accumulator init: bias + input projection (K=5, VALU)
        float xv[8][5];
        #pragma unroll
        for (int r = 0; r < 8; ++r) {
            int m = r + hf * 8;
            #pragma unroll
            for (int j = 0; j < 5; ++j) xv[r][j] = xl[m * 5 + j];
        }
        v8f acc[8];
        #pragma unroll
        for (int s = 0; s < 8; ++s) {
            #pragma unroll
            for (int r = 0; r < 8; ++r) {
                float v = bsum[s];
                #pragma unroll
                for (int j = 0; j < 5; ++j) v = fmaf(xv[r][j], wih[s][j], v);
                acc[s][r] = v;
            }
        }

        // A fragments (h tile 16x256) from LDS
        v16bf afr[8];
        #pragma unroll
        for (int kc = 0; kc < 8; ++kc) {
            const __bf16* p = &hlds[lrow * 256 + kc * 32 + hf * 8];
            v8bf lo = *(const v8bf*)p;
            v8bf hi = *(const v8bf*)(p + 16);
            afr[kc] = __builtin_shufflevector(lo, hi, 0,1,2,3,4,5,6,7,8,9,10,11,12,13,14,15);
        }

        // register-resident half
        #pragma unroll
        for (int kc = 0; kc < 4; ++kc)
            #pragma unroll
            for (int s = 0; s < 8; ++s)
                acc[s] = __builtin_amdgcn_wmma_f32_16x16x32_bf16(
                    false, afr[kc], false, bReg[kc][s], (short)0, acc[s], false, false);

        // streamed half (kc 4..7): opaque zero keeps loads inside the loop
        int tz = 0;
        asm volatile("" : "+v"(tz));
        const __bf16* Ws = WhhS + tz;
        #pragma unroll
        for (int kc = 4; kc < 8; ++kc) {
            v16bf bs[8];
            #pragma unroll
            for (int s = 0; s < 8; ++s)
                bs[s] = *(const v16bf*)&Ws[((size_t)(nt[s] * 8 + kc)) * 512 + lane * 16];
            #pragma unroll
            for (int s = 0; s < 8; ++s)
                acc[s] = __builtin_amdgcn_wmma_f32_16x16x32_bf16(
                    false, afr[kc], false, bs[s], (short)0, acc[s], false, false);
        }

        // gates -> c,h ; write h to y0 (global bf16)
        __bf16 hb[2][8];
        #pragma unroll
        for (int j = 0; j < 2; ++j) {
            #pragma unroll
            for (int r = 0; r < 8; ++r) {
                float iv = sigm  (acc[0 + j][r]);
                float fv = sigm  (acc[2 + j][r]);
                float gv = tanh_f(acc[4 + j][r]);
                float ov = sigm  (acc[6 + j][r]);
                float c  = fv * cst[j][r] + iv * gv;
                cst[j][r] = c;
                float h  = ov * tanh_f(c);
                hb[j][r] = f2bf(h);
                int m   = r + hf * 8;
                int col = (wv << 5) + (j << 4) + lrow;
                y0[((size_t)(bbase + m) * Td + tt) * 512 + dir * 256 + col] = hb[j][r];
            }
        }
        __syncthreads();   // everyone done reading old hlds
        #pragma unroll
        for (int j = 0; j < 2; ++j)
            #pragma unroll
            for (int r = 0; r < 8; ++r)
                hlds[(r + hf * 8) * 256 + (wv << 5) + (j << 4) + lrow] = hb[j][r];
    }
}

// ---------------------------------------------------------------------------
// Layer-1 fw input projection for ALL timesteps as one parallel GEMM:
// xg[row, n] = bias[n] + y0[row,:]·Wih1_fw[n,:], row = b*T+t (M=131072,K=512,N=1024).
// 4 row-tiles per WG so each streamed B fragment is reused 4x.
// ---------------------------------------------------------------------------
__global__ void __launch_bounds__(256, 1) xg_gemm_kernel(
    const __bf16* __restrict__ y0, const __bf16* __restrict__ WihS,
    const float* __restrict__ bih, const float* __restrict__ bhh,
    __bf16* __restrict__ xg)
{
    const int rbase = blockIdx.x * 64;     // 4 row-tiles of 16
    const int tid   = threadIdx.x;
    const int wv    = tid >> 5;
    const int lane  = tid & 31;
    const int hf    = lane >> 4;
    const int lrow  = lane & 15;

    int nt[8], col[8]; float bsum[8];
    #pragma unroll
    for (int s = 0; s < 8; ++s) {
        col[s]  = ((s >> 1) << 8) + (wv << 5) + ((s & 1) << 4) + lrow;
        nt[s]   = 16 * (s >> 1) + 2 * wv + (s & 1);
        bsum[s] = bih[col[s]] + bhh[col[s]];
    }

    v8f acc[4][8];
    #pragma unroll
    for (int rt = 0; rt < 4; ++rt)
        #pragma unroll
        for (int s = 0; s < 8; ++s)
            #pragma unroll
            for (int r = 0; r < 8; ++r) acc[rt][s][r] = bsum[s];

    #pragma unroll
    for (int kc = 0; kc < 16; ++kc) {
        v16bf bs[8];
        #pragma unroll
        for (int s = 0; s < 8; ++s)
            bs[s] = *(const v16bf*)&WihS[((size_t)(nt[s] * 16 + kc)) * 512 + lane * 16];
        #pragma unroll
        for (int rt = 0; rt < 4; ++rt) {
            const __bf16* yr = y0 + (size_t)(rbase + rt * 16 + lrow) * 512 + kc * 32 + hf * 8;
            v8bf lo = *(const v8bf*)yr;
            v8bf hi = *(const v8bf*)(yr + 16);
            v16bf a = __builtin_shufflevector(lo, hi, 0,1,2,3,4,5,6,7,8,9,10,11,12,13,14,15);
            #pragma unroll
            for (int s = 0; s < 8; ++s)
                acc[rt][s] = __builtin_amdgcn_wmma_f32_16x16x32_bf16(
                    false, a, false, bs[s], (short)0, acc[rt][s], false, false);
        }
    }

    #pragma unroll
    for (int rt = 0; rt < 4; ++rt)
        #pragma unroll
        for (int s = 0; s < 8; ++s)
            #pragma unroll
            for (int r = 0; r < 8; ++r)
                xg[(size_t)(rbase + rt * 16 + r + hf * 8) * 1024 + col[s]] = f2bf(acc[rt][s][r]);
}

// ---------------------------------------------------------------------------
// Layer-1 forward scan: acc init from precomputed xg (bias folded in),
// recurrent K=256 chain with hybrid register/streamed Whh (same as layer 0).
// Writes final h (t=T-1) to y1last[:, 0:256].
// ---------------------------------------------------------------------------
__global__ void __launch_bounds__(256, 1) lstm_l1fw_kernel(
    const __bf16* __restrict__ xg, const __bf16* __restrict__ WhhS,
    float* __restrict__ y1last)
{
    __shared__ alignas(16) __bf16 hlds[16 * 256];

    const int bbase = blockIdx.x * 16;
    const int tid   = threadIdx.x;
    const int wv    = tid >> 5;
    const int lane  = tid & 31;
    const int hf    = lane >> 4;
    const int lrow  = lane & 15;

    int nt[8], col[8];
    #pragma unroll
    for (int s = 0; s < 8; ++s) {
        col[s] = ((s >> 1) << 8) + (wv << 5) + ((s & 1) << 4) + lrow;
        nt[s]  = 16 * (s >> 1) + 2 * wv + (s & 1);
    }

    v16bf bReg[4][8];
    #pragma unroll
    for (int kc = 0; kc < 4; ++kc)
        #pragma unroll
        for (int s = 0; s < 8; ++s)
            bReg[kc][s] = *(const v16bf*)&WhhS[((size_t)(nt[s] * 8 + kc)) * 512 + lane * 16];

    for (int i = tid; i < 16 * 256; i += 256) hlds[i] = f2bf(0.f);
    float cst[2][8];
    #pragma unroll
    for (int j = 0; j < 2; ++j)
        #pragma unroll
        for (int r = 0; r < 8; ++r) cst[j][r] = 0.f;
    __syncthreads();

    #pragma unroll 1
    for (int t = 0; t < Td; ++t) {
        __syncthreads();   // prev hlds writes visible

        // acc init from xg (global bf16, t-dependent so naturally in-loop)
        v8f acc[8];
        #pragma unroll
        for (int s = 0; s < 8; ++s)
            #pragma unroll
            for (int r = 0; r < 8; ++r)
                acc[s][r] = bf2f(xg[((size_t)(bbase + r + hf * 8) * Td + t) * 1024 + col[s]]);

        v16bf afr[8];
        #pragma unroll
        for (int kc = 0; kc < 8; ++kc) {
            const __bf16* p = &hlds[lrow * 256 + kc * 32 + hf * 8];
            v8bf lo = *(const v8bf*)p;
            v8bf hi = *(const v8bf*)(p + 16);
            afr[kc] = __builtin_shufflevector(lo, hi, 0,1,2,3,4,5,6,7,8,9,10,11,12,13,14,15);
        }

        #pragma unroll
        for (int kc = 0; kc < 4; ++kc)
            #pragma unroll
            for (int s = 0; s < 8; ++s)
                acc[s] = __builtin_amdgcn_wmma_f32_16x16x32_bf16(
                    false, afr[kc], false, bReg[kc][s], (short)0, acc[s], false, false);

        int tz = 0;
        asm volatile("" : "+v"(tz));
        const __bf16* Ws = WhhS + tz;
        #pragma unroll
        for (int kc = 4; kc < 8; ++kc) {
            v16bf bs[8];
            #pragma unroll
            for (int s = 0; s < 8; ++s)
                bs[s] = *(const v16bf*)&Ws[((size_t)(nt[s] * 8 + kc)) * 512 + lane * 16];
            #pragma unroll
            for (int s = 0; s < 8; ++s)
                acc[s] = __builtin_amdgcn_wmma_f32_16x16x32_bf16(
                    false, afr[kc], false, bs[s], (short)0, acc[s], false, false);
        }

        __bf16 hb[2][8];
        #pragma unroll
        for (int j = 0; j < 2; ++j) {
            #pragma unroll
            for (int r = 0; r < 8; ++r) {
                float iv = sigm  (acc[0 + j][r]);
                float fv = sigm  (acc[2 + j][r]);
                float gv = tanh_f(acc[4 + j][r]);
                float ov = sigm  (acc[6 + j][r]);
                float c  = fv * cst[j][r] + iv * gv;
                cst[j][r] = c;
                float h  = ov * tanh_f(c);
                hb[j][r] = f2bf(h);
                if (t == Td - 1) {
                    int m  = r + hf * 8;
                    int cc = (wv << 5) + (j << 4) + lrow;
                    y1last[(size_t)(bbase + m) * 512 + cc] = h;
                }
            }
        }
        __syncthreads();
        #pragma unroll
        for (int j = 0; j < 2; ++j)
            #pragma unroll
            for (int r = 0; r < 8; ++r)
                hlds[(r + hf * 8) * 256 + (wv << 5) + (j << 4) + lrow] = hb[j][r];
    }
}

// ---------------------------------------------------------------------------
// Layer-1 backward direction = single step at t=T-1 with h=c=0 (the reference's
// reversed scan output at position T-1). Writes y1last[:, 256:512].
// ---------------------------------------------------------------------------
__global__ void __launch_bounds__(256) l1bw_step_kernel(
    const __bf16* __restrict__ y0, const __bf16* __restrict__ WihS,
    const float* __restrict__ bih, const float* __restrict__ bhh,
    float* __restrict__ y1last)
{
    const int bbase = blockIdx.x * 16;
    const int tid   = threadIdx.x;
    const int wv    = tid >> 5;
    const int lane  = tid & 31;
    const int hf    = lane >> 4;
    const int lrow  = lane & 15;

    int nt[8]; float bsum[8];
    #pragma unroll
    for (int s = 0; s < 8; ++s) {
        int col = ((s >> 1) << 8) + (wv << 5) + ((s & 1) << 4) + lrow;
        nt[s]   = 16 * (s >> 1) + 2 * wv + (s & 1);
        bsum[s] = bih[col] + bhh[col];
    }

    v8f acc[8];
    #pragma unroll
    for (int s = 0; s < 8; ++s)
        #pragma unroll
        for (int r = 0; r < 8; ++r) acc[s][r] = bsum[s];

    const __bf16* yr = y0 + ((size_t)(bbase + lrow) * Td + (Td - 1)) * 512;
    #pragma unroll
    for (int kc = 0; kc < 16; ++kc) {
        v8bf lo = *(const v8bf*)(yr + kc * 32 + hf * 8);
        v8bf hi = *(const v8bf*)(yr + kc * 32 + hf * 8 + 16);
        v16bf a = __builtin_shufflevector(lo, hi, 0,1,2,3,4,5,6,7,8,9,10,11,12,13,14,15);
        #pragma unroll
        for (int s = 0; s < 8; ++s) {
            const v16bf b = *(const v16bf*)&WihS[((size_t)(nt[s] * 16 + kc)) * 512 + lane * 16];
            acc[s] = __builtin_amdgcn_wmma_f32_16x16x32_bf16(
                false, a, false, b, (short)0, acc[s], false, false);
        }
    }

    #pragma unroll
    for (int j = 0; j < 2; ++j) {
        #pragma unroll
        for (int r = 0; r < 8; ++r) {
            float iv = sigm  (acc[0 + j][r]);
            float fv = sigm  (acc[2 + j][r]);  (void)fv;  // c_prev = 0
            float gv = tanh_f(acc[4 + j][r]);
            float ov = sigm  (acc[6 + j][r]);
            float c  = iv * gv;
            float h  = ov * tanh_f(c);
            int m  = r + hf * 8;
            int cc = (wv << 5) + (j << 4) + lrow;
            y1last[(size_t)(bbase + m) * 512 + 256 + cc] = h;
        }
    }
}

// ---------------------------------------------------------------------------
// Heads: out = concat(z, mu, sigma_hat), each B*Z f32.
// ---------------------------------------------------------------------------
__global__ void head_kernel(const float* __restrict__ y1last,
                            const float* __restrict__ Wmu,  const float* __restrict__ bmu,
                            const float* __restrict__ Wsig, const float* __restrict__ bsig,
                            const float* __restrict__ eps,  float* __restrict__ out)
{
    int gid = blockIdx.x * blockDim.x + threadIdx.x;
    if (gid >= Bd * Zd) return;
    int b = gid >> 7, z = gid & 127;
    const float* yv = y1last + (size_t)b * 512;
    const float* wm = Wmu  + (size_t)z * 512;
    const float* ws = Wsig + (size_t)z * 512;
    float mu = bmu[z], sg = bsig[z];
    for (int k = 0; k < 512; ++k) {
        float y = yv[k];
        mu = fmaf(y, wm[k], mu);
        sg = fmaf(y, ws[k], sg);
    }
    float zz = mu + __expf(sg) * eps[gid];
    out[gid]               = zz;
    out[Bd * Zd + gid]     = mu;
    out[2 * Bd * Zd + gid] = sg;
}

// ---------------------------------------------------------------------------
extern "C" void kernel_launch(void* const* d_in, const int* in_sizes, int n_in,
                              void* d_out, int out_size, void* d_ws, size_t ws_size,
                              hipStream_t stream) {
    const float* x         = (const float*)d_in[0];
    const float* eps       = (const float*)d_in[1];
    const float* Wih_l0_fw = (const float*)d_in[2];
    const float* Whh_l0_fw = (const float*)d_in[3];
    const float* bih_l0_fw = (const float*)d_in[4];
    const float* bhh_l0_fw = (const float*)d_in[5];
    const float* Wih_l0_bw = (const float*)d_in[6];
    const float* Whh_l0_bw = (const float*)d_in[7];
    const float* bih_l0_bw = (const float*)d_in[8];
    const float* bhh_l0_bw = (const float*)d_in[9];
    const float* Wih_l1_fw = (const float*)d_in[10];
    const float* Whh_l1_fw = (const float*)d_in[11];
    const float* bih_l1_fw = (const float*)d_in[12];
    const float* bhh_l1_fw = (const float*)d_in[13];
    const float* Wih_l1_bw = (const float*)d_in[14];
    /* Whh_l1_bw (d_in[15]) unused: layer-1 bw contributes only its first step (h=0). */
    const float* bih_l1_bw = (const float*)d_in[16];
    const float* bhh_l1_bw = (const float*)d_in[17];
    const float* Wmu       = (const float*)d_in[18];
    const float* bmu       = (const float*)d_in[19];
    const float* Wsigma    = (const float*)d_in[20];
    const float* bsigma    = (const float*)d_in[21];

    char* base = (char*)d_ws;
    size_t off = 0;
    __bf16* y0        = (__bf16*)(base + off); off += (size_t)Bd * Td * 512 * 2;    // 128 MB
    __bf16* xg1       = (__bf16*)(base + off); off += (size_t)Bd * Td * 1024 * 2;   // 256 MB
    __bf16* WhhS0_fw  = (__bf16*)(base + off); off += (size_t)1024 * 256 * 2;
    __bf16* WhhS0_bw  = (__bf16*)(base + off); off += (size_t)1024 * 256 * 2;
    __bf16* WhhS1_fw  = (__bf16*)(base + off); off += (size_t)1024 * 256 * 2;
    __bf16* WihS1_fw  = (__bf16*)(base + off); off += (size_t)1024 * 512 * 2;
    __bf16* WihS1_bw  = (__bf16*)(base + off); off += (size_t)1024 * 512 * 2;
    float*  y1last    = (float*) (base + off); off += (size_t)Bd * 512 * 4;

    {
        int tot = 1024 * 256;
        swizzleB_kernel<<<(tot + 255) / 256, 256, 0, stream>>>(Whh_l0_fw, WhhS0_fw, 256, 8, tot);
        swizzleB_kernel<<<(tot + 255) / 256, 256, 0, stream>>>(Whh_l0_bw, WhhS0_bw, 256, 8, tot);
        swizzleB_kernel<<<(tot + 255) / 256, 256, 0, stream>>>(Whh_l1_fw, WhhS1_fw, 256, 8, tot);
        int tot2 = 1024 * 512;
        swizzleB_kernel<<<(tot2 + 255) / 256, 256, 0, stream>>>(Wih_l1_fw, WihS1_fw, 512, 16, tot2);
        swizzleB_kernel<<<(tot2 + 255) / 256, 256, 0, stream>>>(Wih_l1_bw, WihS1_bw, 512, 16, tot2);
    }

    dim3 grid0(Bd / 16, 2);
    lstm_l0_kernel<<<grid0, 256, 0, stream>>>(
        x, Wih_l0_fw, bih_l0_fw, bhh_l0_fw,
           Wih_l0_bw, bih_l0_bw, bhh_l0_bw,
        WhhS0_fw, WhhS0_bw, y0);

    xg_gemm_kernel<<<(Bd * Td) / 64, 256, 0, stream>>>(
        y0, WihS1_fw, bih_l1_fw, bhh_l1_fw, xg1);

    lstm_l1fw_kernel<<<Bd / 16, 256, 0, stream>>>(xg1, WhhS1_fw, y1last);

    l1bw_step_kernel<<<Bd / 16, 256, 0, stream>>>(
        y0, WihS1_bw, bih_l1_bw, bhh_l1_bw, y1last);

    head_kernel<<<(Bd * Zd + 255) / 256, 256, 0, stream>>>(
        y1last, Wmu, bmu, Wsigma, bsigma, eps, (float*)d_out);
}
